// DualStreamBiMambaBackbone_46248207843467
// MI455X (gfx1250) — matrix operand order repaired and checked by reading, not compiled, not run on hardware
//
#include <hip/hip_runtime.h>
#include <hip/hip_bf16.h>
#include <math.h>

#define BB 64
#define LL 1024
#define DD 128
#define DIC 256      // DI
#define NST 16       // N states
#define RRK 8        // R
#define KCONV 4
#define DBLS 64      // padded row stride of dbl (valid cols: 0..39)
#define TOKN ((size_t)BB * (size_t)LL)   // 65536 tokens

typedef __attribute__((ext_vector_type(16))) _Float16 v16h;
typedef __attribute__((ext_vector_type(8)))  float    v8f;

// ---------------------------------------------------------------- helpers
__device__ __forceinline__ float sigmoidf_(float x) { return 1.f / (1.f + __expf(-x)); }
__device__ __forceinline__ float siluf_(float x)    { return x * sigmoidf_(x); }
__device__ __forceinline__ float geluf_(float x)    { return 0.5f * x * (1.f + erff(x * 0.7071067811865475f)); }
__device__ __forceinline__ float softplusf_(float x){ return (x > 20.f) ? x : log1pf(__expf(x)); }
__device__ __forceinline__ float pe_val(int l, int c) {
  int e = c & ~1;
  float ang = (float)l * __expf(-(float)e * (9.210340371976184f / 128.f)); // ln(1e4)/D
  return (c & 1) ? __cosf(ang) : __sinf(ang);
}

__device__ __forceinline__ float block_sum_128(float v, float* red) {
  int c = threadIdx.x;
  red[c] = v; __syncthreads();
  for (int s = 64; s > 0; s >>= 1) { if (c < s) red[c] += red[c + s]; __syncthreads(); }
  float r = red[0]; __syncthreads();
  return r;
}

// ---------------------------------------------------------------- WMMA GEMM
// One wave (32 threads) per 16x32 output tile (two 16x16 WMMA tiles sharing
// the A fragment). A: [M][K] f32 row-major, Bt: [N][K] f32 (pre-transposed).
// K is compile-time (128 or 256) so the loop fully unrolls and b64 fragment
// loads pipeline across iterations.
// EPI: 0 = +bias, 1 = +residual(C0), 2 = sigmoid(+bias). HB: bias present.
template<int EPI, int K, int HB>
__global__ __launch_bounds__(32)
void gemm_wmma_kernel(const float* __restrict__ A, const float* __restrict__ Bt,
                      const float* __restrict__ bias, const float* __restrict__ C0,
                      float* __restrict__ C, int N) {
  const int lane = threadIdx.x & 31;
  const int g    = lane >> 4;      // half-wave select
  const int r16  = lane & 15;
  const int m0   = blockIdx.y * 16;
  const int n0   = blockIdx.x * 32;
  const float* Arow  = A  + (size_t)(m0 + r16) * K;
  const float* Bcol0 = Bt + (size_t)(n0 + r16) * K;
  const float* Bcol1 = Bcol0 + (size_t)16 * K;
  v8f acc0 = {}, acc1 = {};
#pragma unroll
  for (int k0 = 0; k0 < K; k0 += 32) {
    v16h a, b0, b1;
#pragma unroll
    for (int j = 0; j < 8; ++j) {
      // A 16x32 f16 fragment: VGPR j holds K = 16*g + 2*j + {0,1}
      int ka = k0 + g * 16 + 2 * j;
      float2 av = *(const float2*)(Arow + ka);
      a[2 * j]     = (_Float16)av.x;
      a[2 * j + 1] = (_Float16)av.y;
      // B 32x16 f16 fragment: VGPR j holds K = 16*g + 2*j + {0,1}, N = lane%16
      int kb = k0 + g * 16 + 2 * j;
      float2 bv0 = *(const float2*)(Bcol0 + kb);
      float2 bv1 = *(const float2*)(Bcol1 + kb);
      b0[2 * j]     = (_Float16)bv0.x;
      b0[2 * j + 1] = (_Float16)bv0.y;
      b1[2 * j]     = (_Float16)bv1.x;
      b1[2 * j + 1] = (_Float16)bv1.y;
    }
    acc0 = __builtin_amdgcn_wmma_f32_16x16x32_f16(false, a, false, b0,
                                                  (short)0, acc0, false, false);
    acc1 = __builtin_amdgcn_wmma_f32_16x16x32_f16(false, a, false, b1,
                                                  (short)0, acc1, false, false);
  }
  // epilogue (C/D layout: VGPR rr -> M = rr + 8*(lane/16), N = lane%16)
  const int col0 = n0 + r16;
  const float bv0 = HB ? bias[col0] : 0.f;
  const float bv1 = HB ? bias[col0 + 16] : 0.f;
  size_t idx = (size_t)(m0 + 8 * g) * N + col0;
#pragma unroll
  for (int rr = 0; rr < 8; ++rr, idx += N) {
    float v0 = acc0[rr], v1 = acc1[rr];
    if (EPI == 1) { v0 += C0[idx]; v1 += C0[idx + 16]; }
    else {
      v0 += bv0; v1 += bv1;
      if (EPI == 2) { v0 = sigmoidf_(v0); v1 = sigmoidf_(v1); }
    }
    C[idx] = v0; C[idx + 16] = v1;
  }
}

// Wt[n*K + k] = (n < N) ? W[k*N + n] : 0   (pad to Npad columns)
__global__ void transpose_pad_kernel(const float* __restrict__ W, float* __restrict__ Wt,
                                     int K, int N, int Npad) {
  int i = blockIdx.x * blockDim.x + threadIdx.x;
  if (i >= K * Npad) return;
  int n = i / K, k = i % K;
  Wt[(size_t)n * K + k] = (n < N) ? W[(size_t)k * N + n] : 0.f;
}

// ---------------------------------------------------------------- embedding
__global__ __launch_bounds__(128)
void embed_kernel(const float* __restrict__ x,
                  const float* __restrict__ lw,  const float* __restrict__ lb,
                  const float* __restrict__ lg,  const float* __restrict__ lbt,
                  const float* __restrict__ cw,  const float* __restrict__ cb,
                  const float* __restrict__ cg,  const float* __restrict__ cbt,
                  const float* __restrict__ dir_emb,
                  float* __restrict__ hlen, float* __restrict__ hstruct) {
  __shared__ float red[128];
  size_t tok = blockIdx.x;
  int c = threadIdx.x;
  int l = (int)(tok % LL);
  const float* xp = x + tok * 5;
  float x0 = xp[0], x1 = xp[1], x2 = xp[2], x3 = xp[3], x4 = xp[4];
  float gate = (x4 > 0.5f) ? 1.f : 0.f;
  float pe = pe_val(l, c);
  // length stream: gelu(ln(x[:,0:2]@len_w + len_b)) + pe, * gate
  float v = x0 * lw[c] + x1 * lw[128 + c] + lb[c];
  float m  = block_sum_128(v, red) * (1.f / 128.f);
  float dv = v - m;
  float var = block_sum_128(dv * dv, red) * (1.f / 128.f);
  float hn = dv * rsqrtf(var + 1e-5f) * lg[c] + lbt[c];
  hlen[tok * DD + c] = (geluf_(hn) + pe) * gate;
  // struct stream: gelu(ln(x[:,3:4]@cont_w + cont_b)) + dir_emb + pe, * gate
  float s = x3 * cw[c] + cb[c];
  float ms  = block_sum_128(s, red) * (1.f / 128.f);
  float ds  = s - ms;
  float vs  = block_sum_128(ds * ds, red) * (1.f / 128.f);
  float hs  = ds * rsqrtf(vs + 1e-5f) * cg[c] + cbt[c];
  int didx  = (x2 > 0.f) ? 1 : 0;
  hstruct[tok * DD + c] = (geluf_(hs) + dir_emb[didx * 128 + c] + pe) * gate;
}

// ---------------------------------------------------------------- layernorm
__global__ __launch_bounds__(128)
void ln_kernel(const float* __restrict__ h, const float* __restrict__ g,
               const float* __restrict__ b, float* __restrict__ out) {
  __shared__ float red[128];
  size_t tok = blockIdx.x;
  int c = threadIdx.x;
  float v = h[tok * DD + c];
  float m  = block_sum_128(v, red) * (1.f / 128.f);
  float dv = v - m;
  float var = block_sum_128(dv * dv, red) * (1.f / 128.f);
  out[tok * DD + c] = dv * rsqrtf(var + 1e-5f) * g[c] + b[c];
}

// ---------------------------------------------------------------- mamba pieces
// u = silu(causal_dwconv(xz[..., :256])) ; z stays in xz[..., 256:]
__global__ void dwconv_silu_kernel(const float* __restrict__ xz, const float* __restrict__ cw,
                                   const float* __restrict__ cb, float* __restrict__ u) {
  size_t i = (size_t)blockIdx.x * blockDim.x + threadIdx.x;
  if (i >= TOKN * DIC) return;
  int c = (int)(i % DIC);
  size_t tok = i / DIC;
  int l = (int)(tok % LL);
  size_t b0 = tok - l;
  float acc = cb[c];
#pragma unroll
  for (int k = 0; k < KCONV; ++k) {
    int li = l - (KCONV - 1) + k;
    if (li >= 0) acc += cw[c * KCONV + k] * xz[(b0 + li) * 512 + c];
  }
  u[i] = siluf_(acc);
}

// dt = softplus(dbl[:, :8] @ dt_w + dt_b)
__global__ void dt_kernel(const float* __restrict__ dbl, const float* __restrict__ dtw,
                          const float* __restrict__ dtb, float* __restrict__ dt) {
  size_t i = (size_t)blockIdx.x * blockDim.x + threadIdx.x;
  if (i >= TOKN * DIC) return;
  int c = (int)(i % DIC);
  size_t tok = i / DIC;
  const float* dr = dbl + tok * DBLS;
  float s = dtb[c];
#pragma unroll
  for (int r = 0; r < RRK; ++r) s += dr[r] * dtw[r * DIC + c];
  dt[i] = softplusf_(s);
}

// selective scan: one thread per (b, d); B in dbl[...,8:24], C in dbl[...,24:40]
__global__ __launch_bounds__(256)
void scan_kernel(const float* __restrict__ u, const float* __restrict__ dt,
                 const float* __restrict__ dbl, const float* __restrict__ Alog,
                 const float* __restrict__ Dd, float* __restrict__ y) {
  int i = blockIdx.x * blockDim.x + threadIdx.x;
  if (i >= BB * DIC) return;
  int d = i % DIC, b = i / DIC;
  float A[NST], h[NST];
#pragma unroll
  for (int n = 0; n < NST; ++n) { A[n] = -__expf(Alog[d * NST + n]); h[n] = 0.f; }
  float Dc = Dd[d];
  size_t base = (size_t)b * LL;
  for (int t = 0; t < LL; ++t) {
    size_t tok = base + t;
    float dtv = dt[tok * DIC + d];
    float uv  = u[tok * DIC + d];
    const float* br = dbl + tok * DBLS;
    float du = dtv * uv;
    float acc = 0.f;
#pragma unroll
    for (int n = 0; n < NST; ++n) {
      h[n] = __expf(dtv * A[n]) * h[n] + du * br[8 + n];
      acc += h[n] * br[24 + n];
    }
    y[tok * DIC + d] = acc + uv * Dc;
  }
}

// y *= silu(z) where z = xz[..., 256:]
__global__ void zgate_kernel(float* __restrict__ y, const float* __restrict__ xz) {
  size_t i = (size_t)blockIdx.x * blockDim.x + threadIdx.x;
  if (i >= TOKN * DIC) return;
  size_t tok = i / DIC; int c = (int)(i % DIC);
  y[i] *= siluf_(xz[tok * 512 + 256 + c]);
}

// ---------------------------------------------------------------- glue
__global__ void reverse_kernel(const float* __restrict__ in, float* __restrict__ out) {
  size_t i = (size_t)blockIdx.x * blockDim.x + threadIdx.x;
  if (i >= TOKN * DD) return;
  int c = (int)(i % DD);
  size_t tok = i / DD;
  int l = (int)(tok % LL);
  size_t b0 = tok - l;
  out[i] = in[(b0 + (LL - 1 - l)) * DD + c];
}

// hcat = [hf, reverse(hb)]
__global__ void concat_bidir_kernel(const float* __restrict__ hf, const float* __restrict__ hb,
                                    float* __restrict__ out) {
  size_t i = (size_t)blockIdx.x * blockDim.x + threadIdx.x;
  if (i >= TOKN * 256) return;
  int c = (int)(i % 256);
  size_t tok = i / 256;
  int l = (int)(tok % LL);
  size_t b0 = tok - l;
  out[i] = (c < 128) ? hf[tok * 128 + c]
                     : hb[(b0 + (LL - 1 - l)) * 128 + (c - 128)];
}

__global__ void concat2_kernel(const float* __restrict__ a, const float* __restrict__ b,
                               float* __restrict__ out) {
  size_t i = (size_t)blockIdx.x * blockDim.x + threadIdx.x;
  if (i >= TOKN * 256) return;
  int c = (int)(i % 256);
  size_t tok = i / 256;
  out[i] = (c < 128) ? a[tok * 128 + c] : b[tok * 128 + (c - 128)];
}

__global__ void mul_kernel(float* __restrict__ a, const float* __restrict__ b) {
  size_t i = (size_t)blockIdx.x * blockDim.x + threadIdx.x;
  if (i >= TOKN * 256) return;
  a[i] *= b[i];
}

// masked mean over L
__global__ __launch_bounds__(256)
void reduce_kernel(const float* __restrict__ zs, const float* __restrict__ x,
                   float* __restrict__ out) {
  int b = blockIdx.x, c = threadIdx.x;
  float acc = 0.f, cnt = 0.f;
  for (int l = 0; l < LL; ++l) {
    size_t tok = (size_t)b * LL + l;
    float g = (x[tok * 5 + 4] > 0.5f) ? 1.f : 0.f;
    acc += g * zs[tok * 256 + c];
    cnt += g;
  }
  out[b * 256 + c] = acc / fmaxf(cnt, 1.f);
}

// ---------------------------------------------------------------- host
extern "C" void kernel_launch(void* const* d_in, const int* in_sizes, int n_in,
                              void* d_out, int out_size, void* d_ws, size_t ws_size,
                              hipStream_t stream) {
  (void)in_sizes; (void)n_in; (void)out_size; (void)ws_size;
  const float* x        = (const float*)d_in[0];
  const float* len_w    = (const float*)d_in[1];
  const float* len_b    = (const float*)d_in[2];
  const float* len_ln_g = (const float*)d_in[3];
  const float* len_ln_b = (const float*)d_in[4];
  const float* cont_w   = (const float*)d_in[5];
  const float* cont_b   = (const float*)d_in[6];
  const float* cont_ln_g= (const float*)d_in[7];
  const float* cont_ln_b= (const float*)d_in[8];
  const float* dir_emb  = (const float*)d_in[9];
  const float* m_in_w   = (const float*)d_in[10];
  const float* m_conv_w = (const float*)d_in[11];
  const float* m_conv_b = (const float*)d_in[12];
  const float* m_xproj_w= (const float*)d_in[13];
  const float* m_dt_w   = (const float*)d_in[14];
  const float* m_dt_b   = (const float*)d_in[15];
  const float* m_Alog   = (const float*)d_in[16];
  const float* m_D      = (const float*)d_in[17];
  const float* m_out_w  = (const float*)d_in[18];
  const float* m_norm_g = (const float*)d_in[19];
  const float* m_norm_b = (const float*)d_in[20];
  const float* lenproj_w   = (const float*)d_in[21];
  const float* lenproj_b   = (const float*)d_in[22];
  const float* structproj_w= (const float*)d_in[23];
  const float* structproj_b= (const float*)d_in[24];
  const float* gate_w   = (const float*)d_in[25];
  const float* gate_b   = (const float*)d_in[26];
  const float* fuse_w   = (const float*)d_in[27];
  const float* fuse_b   = (const float*)d_in[28];
  const float* proj_w   = (const float*)d_in[29];
  const float* proj_b   = (const float*)d_in[30];

  // ---- workspace layout (floats, bump allocator)
  float* w = (float*)d_ws;
  size_t off = 0;
  auto alloc = [&](size_t n) { float* p = w + off; off += n; return p; };
  float* hlen    = alloc(TOKN * DD);
  float* hstruct = alloc(TOKN * DD);
  float* soutL   = alloc(TOKN * DD);
  float* soutS   = alloc(TOKN * DD);
  float* hbb     = alloc(TOKN * DD);        // reversed-stream working buffer
  float* hn      = alloc(TOKN * DD);
  float* xz      = alloc(TOKN * 512);
  float* ub      = alloc(TOKN * DIC);
  float* dblb    = alloc(TOKN * DBLS);
  float* dtbuf   = alloc(TOKN * DIC);
  float* yb      = alloc(TOKN * DIC);
  float* hcat    = alloc(TOKN * 256);
  float* Wt      = alloc(512 * 256);        // transposed-weight scratch

  const int n128 = (int)((TOKN * DD  + 255) / 256);
  const int n256i= (int)((TOKN * DIC + 255) / 256);
  const int nc256= (int)((TOKN * 256 + 255) / 256);

  // epi: 0 = +bias (HB from bias!=null), 1 = residual, 2 = sigmoid(+bias)
  auto gemm = [&](const float* A, const float* W_, int K, int N, int Npad,
                  const float* bias, const float* C0, float* C, int epi) {
    transpose_pad_kernel<<<(K * Npad + 255) / 256, 256, 0, stream>>>(W_, Wt, K, N, Npad);
    dim3 grid(Npad / 32, (unsigned)(TOKN / 16));
    if (K == 128) {
      gemm_wmma_kernel<0, 128, 0><<<grid, 32, 0, stream>>>(A, Wt, nullptr, nullptr, C, Npad);
    } else {  // K == 256
      if (epi == 1)
        gemm_wmma_kernel<1, 256, 0><<<grid, 32, 0, stream>>>(A, Wt, nullptr, C0, C, Npad);
      else if (epi == 2)
        gemm_wmma_kernel<2, 256, 1><<<grid, 32, 0, stream>>>(A, Wt, bias, nullptr, C, Npad);
      else if (bias)
        gemm_wmma_kernel<0, 256, 1><<<grid, 32, 0, stream>>>(A, Wt, bias, nullptr, C, Npad);
      else
        gemm_wmma_kernel<0, 256, 0><<<grid, 32, 0, stream>>>(A, Wt, nullptr, nullptr, C, Npad);
    }
  };

  auto run_layer = [&](float* h, int li) {
    ln_kernel<<<(unsigned)TOKN, 128, 0, stream>>>(h, m_norm_g + li * DD, m_norm_b + li * DD, hn);
    gemm(hn, m_in_w + (size_t)li * DD * 2 * DIC, DD, 2 * DIC, 2 * DIC, nullptr, nullptr, xz, 0);
    dwconv_silu_kernel<<<n256i, 256, 0, stream>>>(xz, m_conv_w + li * DIC * KCONV,
                                                  m_conv_b + li * DIC, ub);
    gemm(ub, m_xproj_w + (size_t)li * DIC * 40, DIC, 40, DBLS, nullptr, nullptr, dblb, 0);
    dt_kernel<<<n256i, 256, 0, stream>>>(dblb, m_dt_w + li * RRK * DIC, m_dt_b + li * DIC, dtbuf);
    scan_kernel<<<(BB * DIC) / 256, 256, 0, stream>>>(ub, dtbuf, dblb,
                                                      m_Alog + li * DIC * NST, m_D + li * DIC, yb);
    zgate_kernel<<<n256i, 256, 0, stream>>>(yb, xz);
    gemm(yb, m_out_w + (size_t)li * DIC * DD, DIC, DD, DD, nullptr, h, h, 1);  // h += y@out_w
  };

  auto run_bidir = [&](float* h, int f0, int f1, int b0, int b1,
                       const float* pw, const float* pb, float* outb) {
    reverse_kernel<<<n128, 256, 0, stream>>>(h, hbb);
    run_layer(h, f0);   run_layer(h, f1);
    run_layer(hbb, b0); run_layer(hbb, b1);
    concat_bidir_kernel<<<nc256, 256, 0, stream>>>(h, hbb, hcat);
    gemm(hcat, pw, 256, DD, DD, pb, nullptr, outb, 0);
  };

  // ---- embedding
  embed_kernel<<<(unsigned)TOKN, 128, 0, stream>>>(x, len_w, len_b, len_ln_g, len_ln_b,
                                                   cont_w, cont_b, cont_ln_g, cont_ln_b,
                                                   dir_emb, hlen, hstruct);
  // ---- two bidirectional streams (shared 8-layer parameter bank)
  run_bidir(hlen,    0, 1, 2, 3, lenproj_w,    lenproj_b,    soutL);
  run_bidir(hstruct, 4, 5, 6, 7, structproj_w, structproj_b, soutS);

  // ---- fusion head (reuse freed scratch buffers)
  float* fused = hcat;  float* gbuf = ub;  float* ff = yb;  float* zseq = dtbuf;
  concat2_kernel<<<nc256, 256, 0, stream>>>(soutL, soutS, fused);
  gemm(fused, gate_w, 256, 256, 256, gate_b, nullptr, gbuf, 2);   // sigmoid gate
  gemm(fused, fuse_w, 256, 256, 256, fuse_b, nullptr, ff, 0);
  mul_kernel<<<nc256, 256, 0, stream>>>(ff, gbuf);
  gemm(ff, proj_w, 256, 256, 256, proj_b, nullptr, zseq, 0);
  reduce_kernel<<<BB, 256, 0, stream>>>(zseq, x, (float*)d_out);
}
